// PCDEncoder_16209206575259
// MI455X (gfx1250) — matrix operand order repaired and checked by reading, not compiled, not run on hardware
//
#include <hip/hip_runtime.h>
#include <math.h>

typedef _Float16 v16h __attribute__((ext_vector_type(16)));
typedef _Float16 v8h  __attribute__((ext_vector_type(8)));
typedef float    v8f  __attribute__((ext_vector_type(8)));

#define NB    8
#define NPTS  131072
#define VOX   343          // 7^3
#define BD    64

// ---------------- workspace layout (bytes) ----------------
#define OFF_RED     0u            // 8*6 uints (min/max encoded)          192 B
#define OFF_PARAMS  256u          // 8*12 floats: lo[3],hi[3],ab[6]       384 B
#define OFF_FRAG    1024u         // 20 frags * 32 lanes * 16 halves    20480 B
#define OFF_GRID    21760u        // 8*343*64 f32                      702464 B
#define OFF_VALID   724224u       // 8*343 uint                         10976 B
#define OFF_XVX     735232u       // 8*343*74 f32                      812224 B
#define OFF_E0      1547520u      // 8*125*64 f32                      256000 B
#define OFF_E1      1803520u      // 8*27*64 f32                        55296 B
#define OFF_E2      1858816u      // 8*27*64 f32                        55296 B
#define OFF_D0      1914112u      // 8*27*64 f32                        55296 B
#define OFF_D1      1969408u      // 8*125*64 f32                      256000 B
#define OFF_D2      2225408u      // 8*343*64 f32                      702464 B
#define OFF_G       2927872u      // 8*64 f32                            2048 B

// output layout (floats)
#define OUT_VALID   0
#define OUT_G       2744
#define OUT_FEAT    3256
#define OUT_AABB    91064
#define OUT_ORIG    91112

__device__ __forceinline__ unsigned encf(float f) {
  unsigned u = __float_as_uint(f);
  return (u & 0x80000000u) ? ~u : (u | 0x80000000u);
}
__device__ __forceinline__ float decf(unsigned u) {
  return __uint_as_float((u & 0x80000000u) ? (u & 0x7FFFFFFFu) : ~u);
}

// ---------------- init: zero voxel grid / flags, seed reducers -------------
__global__ void init_ws_kernel(float* grid, unsigned* vflag, unsigned* red) {
  int t = blockIdx.x * 256 + threadIdx.x;
  if (t < NB * VOX * BD) grid[t] = 0.0f;
  if (t < NB * VOX)      vflag[t] = 0u;
  if (t < NB * 6)        red[t] = ((t % 6) < 3) ? 0xFFFFFFFFu : 0u;
}

// ---------------- weight fragment prep: W^T in WMMA **A** layout -----------
// D = A x B with A = W^T (16 out-ch x 32 in-ch), B = X^T (32 in-ch x 16 pts).
// A 16x32 f16 k order used here (contiguous): lane<16:  M=lane,    K=kb+h
//                     lane>=16: M=lane-16, K=kb+16+h
//   i.e. K = kb + h + (lane>=16 ? 16 : 0)
// frag f: layer0 ct=f (f<4, kb=0, row9 = bias); layer1 f=4+c*4+ct; layer2 f=12+c*4+ct
__global__ void prep_frags_kernel(const float* w0, const float* b0,
                                  const float* w1, const float* w2,
                                  _Float16* frag) {
  int t = blockIdx.x * 256 + threadIdx.x;
  if (t >= 20 * 512) return;
  int f = t / 512, r = t % 512, lane = r / 16, h = r % 16;
  int layer, c, ct;
  if (f < 4)       { layer = 0; c = 0;            ct = f; }
  else if (f < 12) { layer = 1; c = (f - 4) / 4;  ct = (f - 4) % 4; }
  else             { layer = 2; c = (f - 12) / 4; ct = (f - 12) % 4; }
  int K = c * 32 + h + ((lane >= 16) ? 16 : 0);
  int n = ct * 16 + (lane & 15);      // output channel
  float v = 0.0f;
  if (layer == 0) {
    if (K < 9)       v = w0[K * BD + n];
    else if (K == 9) v = b0[n];       // bias folded: input K=9 is constant 1
  } else if (layer == 1) v = w1[K * BD + n];
  else                   v = w2[K * BD + n];
  frag[(f * 32 + lane) * 16 + h] = (_Float16)v;
}

// ---------------- per-batch min/max reduction ------------------------------
__global__ __launch_bounds__(256) void reduce_minmax_kernel(const float* xyz, unsigned* red) {
  int b = blockIdx.y;
  float mn[3] = { 1e30f, 1e30f, 1e30f };
  float mx[3] = { -1e30f, -1e30f, -1e30f };
  for (int i = blockIdx.x * 256 + threadIdx.x; i < NPTS; i += gridDim.x * 256) {
    const float* p = xyz + ((size_t)b * NPTS + i) * 3;
    for (int a = 0; a < 3; a++) { float v = p[a]; mn[a] = fminf(mn[a], v); mx[a] = fmaxf(mx[a], v); }
  }
  __shared__ unsigned smin[3], smax[3];
  if (threadIdx.x < 3) { smin[threadIdx.x] = 0xFFFFFFFFu; smax[threadIdx.x] = 0u; }
  __syncthreads();
  for (int a = 0; a < 3; a++) {
    atomicMin(&smin[a], encf(mn[a]));
    atomicMax(&smax[a], encf(mx[a]));
  }
  __syncthreads();
  if (threadIdx.x < 3) {
    atomicMin(&red[b * 6 + threadIdx.x],     smin[threadIdx.x]);
    atomicMax(&red[b * 6 + 3 + threadIdx.x], smax[threadIdx.x]);
  }
}

// ---------------- finalize aabb (replicates reference's repeat quirk) ------
__global__ void finalize_aabb_kernel(const unsigned* red, const float* ls,
                                     float* params, float* out_aabb, float* out_orig) {
  int b = threadIdx.x;
  if (b >= NB) return;
  float amin[3], amax[3], c[3];
  float ml = -1e30f;
  for (int a = 0; a < 3; a++) {
    amin[a] = decf(red[b * 6 + a]) - 0.01f;
    amax[a] = decf(red[b * 6 + 3 + a]) + 0.01f;
    c[a] = 0.5f * (amin[a] + amax[a]);
    ml = fmaxf(ml, amax[a] - amin[a]);
  }
  float h = 0.5f * ml;
  // aabb = half + repeat(center,2): [c0-h, c0-h, c1-h, c1+h, c2+h, c2+h]
  float a6[6] = { c[0]-h, c[0]-h, c[1]-h, c[1]+h, c[2]+h, c[2]+h };
  float e = expf(ls[0]);
  float* P = params + b * 12;
  for (int a = 0; a < 6; a++) { P[a] = a6[a]; out_aabb[b * 6 + a] = a6[a]; }
  for (int a = 0; a < 3; a++) {
    P[6 + a]     = e * amin[a];
    P[6 + 3 + a] = e * amax[a];
    out_orig[b * 6 + a]     = amin[a];
    out_orig[b * 6 + 3 + a] = amax[a];
  }
}

// ---------------- WMMA builtin wrapper -------------------------------------
__device__ __forceinline__ v8f wmma_f16(v16h a, v16h b, v8f c) {
  return __builtin_amdgcn_wmma_f32_16x16x32_f16(false, a, false, b, (short)0, c, false, false);
}

// D (point-major) -> next-layer B fragments via lane<->lane^16 swap.
// Shuffles are unconditional for all lanes (inactive-lane reads return 0).
__device__ __forceinline__ void make_bfrags(const v8f d[4], v16h bn[2], bool hiHalf) {
  union U { v8h h; unsigned u[4]; };
  U own[4], oth[4];
#pragma unroll
  for (int t = 0; t < 4; t++) {
#pragma unroll
    for (int r = 0; r < 8; r++) own[t].h[r] = (_Float16)d[t][r];
#pragma unroll
    for (int i = 0; i < 4; i++) oth[t].u[i] = __shfl_xor(own[t].u[i], 16, 32);
  }
#pragma unroll
  for (int c = 0; c < 2; c++) {
#pragma unroll
    for (int r = 0; r < 8; r++) {
      bn[c][r]     = hiHalf ? oth[2 * c + 1].h[r] : own[2 * c].h[r];
      bn[c][8 + r] = hiHalf ? own[2 * c + 1].h[r] : oth[2 * c].h[r];
    }
  }
}

// ---------------- point MLP (WMMA, weights as A) + segment-max scatter -----
__global__ __launch_bounds__(256) void mlp_scatter_kernel(
    const float* __restrict__ xyz, const float* __restrict__ normal,
    const _Float16* __restrict__ frag,
    const float* __restrict__ b1, const float* __restrict__ b2,
    const float* __restrict__ params, float* grid, unsigned* vflag) {
  const int lane   = threadIdx.x & 31;
  const int wave   = blockIdx.x * 8 + (threadIdx.x >> 5);
  const int nwave  = gridDim.x * 8;
  const int nlo    = lane & 15;
  const int rbase  = (lane >> 4) * 8;
  const bool hiHalf = lane >= 16;

  // register-resident weight fragments (20 x v16h)
  v16h aw[20];
#pragma unroll
  for (int f = 0; f < 20; f++)
    aw[f] = *(const v16h*)(frag + (size_t)(f * 32 + lane) * 16);

  // loop-invariant per-lane bias C-initializers (layer0 bias folded into W)
  v8f bias1v[4], bias2v[4];
#pragma unroll
  for (int ct = 0; ct < 4; ct++) {
    const float* p1 = b1 + ct * 16 + rbase;
    const float* p2 = b2 + ct * 16 + rbase;
#pragma unroll
    for (int r = 0; r < 8; r++) { bias1v[ct][r] = p1[r]; bias2v[ct][r] = p2[r]; }
  }

  const int TILES = NB * (NPTS / 16);
  for (int t = wave; t < TILES; t += nwave) {
    int b  = t / (NPTS / 16);
    int p0 = (t % (NPTS / 16)) * 16;
    size_t pidx = ((size_t)b * NPTS + p0 + nlo) * 3;
    float px = xyz[pidx], py = xyz[pidx + 1], pz = xyz[pidx + 2];
    float nx = normal[pidx], ny = normal[pidx + 1], nz = normal[pidx + 2];
    const float* P = params + b * 12;
    float sx = (px - P[0]) / (P[3] - P[0]) * 7.0f;
    float sy = (py - P[1]) / (P[4] - P[1]) * 7.0f;
    float sz = (pz - P[2]) / (P[5] - P[2]) * 7.0f;
    float fx = floorf(sx), fy = floorf(sy), fz = floorf(sz);
    int idxflat = 49 * (int)fx + 7 * (int)fy + (int)fz;
    bool inRange = (idxflat >= 0) && (idxflat < VOX);

    // B0 = X^T: lane<16 holds K=0..15 of its point (dims 0..8 live, 9 == 1.0
    // for the folded bias), lanes>=16 hold K=16..31 == 0.
    float in10[10] = { px, py, pz,
                       (sx - fx) * 2.0f - 1.0f, (sy - fy) * 2.0f - 1.0f, (sz - fz) * 2.0f - 1.0f,
                       nx, ny, nz, 1.0f };
    v16h bf;
#pragma unroll
    for (int h = 0; h < 16; h++) bf[h] = (_Float16)0.0f;
    if (!hiHalf) {
#pragma unroll
      for (int h = 0; h < 10; h++) bf[h] = (_Float16)in10[h];
    }

    v8f zc;
#pragma unroll
    for (int i = 0; i < 8; i++) zc[i] = 0.0f;

    // ---- layer 0 (bias folded) ----
    v8f d[4], skip[4];
#pragma unroll
    for (int ct = 0; ct < 4; ct++) {
      v8f acc = wmma_f16(aw[ct], bf, zc);
#pragma unroll
      for (int i = 0; i < 8; i++) acc[i] = fmaxf(acc[i], 0.0f);
      d[ct] = acc;
      skip[ct] = acc;
    }
    v16h bn[2];
    make_bfrags(d, bn, hiHalf);

    // ---- layer 1 ----
#pragma unroll
    for (int ct = 0; ct < 4; ct++) {
      v8f acc = bias1v[ct];
      acc = wmma_f16(aw[4 + ct], bn[0], acc);
      acc = wmma_f16(aw[8 + ct], bn[1], acc);
#pragma unroll
      for (int i = 0; i < 8; i++) acc[i] = fmaxf(acc[i], 0.0f);
      d[ct] = acc;
    }
    make_bfrags(d, bn, hiHalf);

    // ---- layer 2 + skip + scatter (lane already owns its point's idx) ----
    if (!hiHalf && inRange) vflag[b * VOX + idxflat] = 1u;
    float* gbase = grid + ((size_t)b * VOX + (inRange ? idxflat : 0)) * BD;
#pragma unroll
    for (int ct = 0; ct < 4; ct++) {
      v8f acc = bias2v[ct];
      acc = wmma_f16(aw[12 + ct], bn[0], acc);
      acc = wmma_f16(aw[16 + ct], bn[1], acc);
      int ch = ct * 16 + rbase;
#pragma unroll
      for (int r = 0; r < 8; r++) {
        float v = fmaxf(acc[r], 0.0f) + skip[ct][r];  // x >= 0 guaranteed
        if (inRange)
          atomicMax((unsigned*)(gbase + ch + r), __float_as_uint(v));
      }
    }
  }
}

// ---------------- assemble x_vx (B,343,74) + valid output ------------------
__global__ void voxel_assemble_kernel(const float* grid, const unsigned* vflag,
                                      const float* params, float* xvx, float* out_valid) {
  int t = blockIdx.x * 256 + threadIdx.x;
  if (t >= NB * VOX * 74) return;
  int ch = t % 74, v = (t / 74) % VOX, b = t / (74 * VOX);
  float valid = vflag[b * VOX + v] ? 1.0f : 0.0f;
  float val;
  if (ch == 0) { val = valid; out_valid[b * VOX + v] = valid; }
  else if (ch < 65) val = grid[((size_t)b * VOX + v) * BD + (ch - 1)];
  else if (ch < 68) {
    int a = ch - 65;
    int ia = (a == 0) ? (v / 49) : (a == 1) ? ((v / 7) % 7) : (v % 7);
    const float* P = params + b * 12;
    val = ((float)ia + 0.5f) / 7.0f * (P[3 + a] - P[a]) + P[a];
  } else val = params[b * 12 + 6 + (ch - 68)];
  xvx[t] = val;
}

// ---------------- conv3x3x3 (unflipped, pad p) + bias + relu + LayerNorm ---
__global__ __launch_bounds__(64) void conv3d_ln_kernel(
    const float* __restrict__ inA, const float* __restrict__ inB, int cinA, int cinB,
    int Din, int Dout, int pad,
    const float* __restrict__ krn, const float* __restrict__ bias,
    const float* __restrict__ gamma, const float* __restrict__ beta,
    float* __restrict__ out) {
  int D3o = Dout * Dout * Dout;
  int b = blockIdx.x / D3o, s = blockIdx.x % D3o;
  int o0 = s / (Dout * Dout), o1 = (s / Dout) % Dout, o2 = s % Dout;
  int oc = threadIdx.x;
  int cin = cinA + cinB;
  int D3i = Din * Din * Din;
  float acc = 0.0f;
  for (int k0 = 0; k0 < 3; k0++) {
    int i0 = o0 + k0 - pad; if ((unsigned)i0 >= (unsigned)Din) continue;
    for (int k1 = 0; k1 < 3; k1++) {
      int i1 = o1 + k1 - pad; if ((unsigned)i1 >= (unsigned)Din) continue;
      for (int k2 = 0; k2 < 3; k2++) {
        int i2 = o2 + k2 - pad; if ((unsigned)i2 >= (unsigned)Din) continue;
        int sin = (i0 * Din + i1) * Din + i2;
        const float* kb = krn + ((size_t)((k0 * 3 + k1) * 3 + k2) * cin) * BD + oc;
        const float* ia = inA + ((size_t)b * D3i + sin) * cinA;
        for (int ci = 0; ci < cinA; ci++) acc += ia[ci] * kb[(size_t)ci * BD];
        if (cinB) {
          const float* ib = inB + ((size_t)b * D3i + sin) * cinB;
          for (int ci = 0; ci < cinB; ci++) acc += ib[ci] * kb[(size_t)(cinA + ci) * BD];
        }
      }
    }
  }
  float tv = fmaxf(acc + bias[oc], 0.0f);
  __shared__ float sv[BD];
  sv[oc] = tv;
  __syncthreads();
  float mean = 0.0f;
  for (int i = 0; i < BD; i++) mean += sv[i];
  mean *= (1.0f / BD);
  float var = 0.0f;
  for (int i = 0; i < BD; i++) { float dv = sv[i] - mean; var += dv * dv; }
  var *= (1.0f / BD);
  out[((size_t)b * D3o + s) * BD + oc] =
      (tv - mean) * rsqrtf(var + 1e-6f) * gamma[oc] + beta[oc];
}

// ---------------- global mean of e2 over 27 positions ----------------------
__global__ void gmean_kernel(const float* e2, float* gbuf, float* out_g) {
  int t = blockIdx.x * 64 + threadIdx.x;
  if (t >= NB * BD) return;
  int b = t / BD, c = t % BD;
  float s = 0.0f;
  for (int p = 0; p < 27; p++) s += e2[((size_t)b * 27 + p) * BD + c];
  s *= (1.0f / 27.0f);
  gbuf[t] = s;
  out_g[t] = s;
}

// ---------------- final mix (128->64 LN) and feat (64->32) -----------------
__global__ __launch_bounds__(64) void mix_feat_kernel(
    const float* __restrict__ d2, const float* __restrict__ gbuf,
    const float* __restrict__ wm, const float* __restrict__ bm,
    const float* __restrict__ gm, const float* __restrict__ hm,
    const float* __restrict__ wo, const float* __restrict__ bo,
    float* __restrict__ out_feat) {
  int b = blockIdx.x / VOX, s = blockIdx.x % VOX;
  int c = threadIdx.x;
  const float* din = d2 + ((size_t)b * VOX + s) * BD;
  float acc = 0.0f;
  for (int k = 0; k < BD; k++) acc += din[k] * wm[(size_t)k * BD + c];
  for (int k = 0; k < BD; k++) acc += gbuf[b * BD + k] * wm[(size_t)(BD + k) * BD + c];
  float tv = fmaxf(acc + bm[c], 0.0f);
  __shared__ float sv[BD];
  __shared__ float sl[BD];
  sv[c] = tv;
  __syncthreads();
  float mean = 0.0f;
  for (int i = 0; i < BD; i++) mean += sv[i];
  mean *= (1.0f / BD);
  float var = 0.0f;
  for (int i = 0; i < BD; i++) { float dv = sv[i] - mean; var += dv * dv; }
  var *= (1.0f / BD);
  sl[c] = (tv - mean) * rsqrtf(var + 1e-6f) * gm[c] + hm[c];
  __syncthreads();
  if (c < 32) {
    float f = bo[c];
    for (int k = 0; k < BD; k++) f += sl[k] * wo[(size_t)k * 32 + c];
    out_feat[((size_t)b * VOX + s) * 32 + c] = f;
  }
}

// ===========================================================================
extern "C" void kernel_launch(void* const* d_in, const int* in_sizes, int n_in,
                              void* d_out, int out_size, void* d_ws, size_t ws_size,
                              hipStream_t stream) {
  const float* xyz    = (const float*)d_in[0];
  const float* normal = (const float*)d_in[1];
  const float* ls     = (const float*)d_in[2];
  const float* w0 = (const float*)d_in[3];  const float* b0 = (const float*)d_in[4];
  const float* w1 = (const float*)d_in[5];  const float* b1 = (const float*)d_in[6];
  const float* w2 = (const float*)d_in[7];  const float* b2 = (const float*)d_in[8];
  const float* ke0 = (const float*)d_in[9];  const float* be0 = (const float*)d_in[10];
  const float* ge0 = (const float*)d_in[11]; const float* he0 = (const float*)d_in[12];
  const float* ke1 = (const float*)d_in[13]; const float* be1 = (const float*)d_in[14];
  const float* ge1 = (const float*)d_in[15]; const float* he1 = (const float*)d_in[16];
  const float* ke2 = (const float*)d_in[17]; const float* be2 = (const float*)d_in[18];
  const float* ge2 = (const float*)d_in[19]; const float* he2 = (const float*)d_in[20];
  const float* kd0 = (const float*)d_in[21]; const float* bd0 = (const float*)d_in[22];
  const float* gd0 = (const float*)d_in[23]; const float* hd0 = (const float*)d_in[24];
  const float* kd1 = (const float*)d_in[25]; const float* bd1 = (const float*)d_in[26];
  const float* gd1 = (const float*)d_in[27]; const float* hd1 = (const float*)d_in[28];
  const float* kd2 = (const float*)d_in[29]; const float* bd2 = (const float*)d_in[30];
  const float* gd2 = (const float*)d_in[31]; const float* hd2 = (const float*)d_in[32];
  const float* wm = (const float*)d_in[33];  const float* bm = (const float*)d_in[34];
  const float* gm = (const float*)d_in[35];  const float* hm = (const float*)d_in[36];
  const float* wo = (const float*)d_in[37];  const float* bo = (const float*)d_in[38];

  char* ws = (char*)d_ws;
  unsigned* red   = (unsigned*)(ws + OFF_RED);
  float* params   = (float*)(ws + OFF_PARAMS);
  _Float16* frag  = (_Float16*)(ws + OFF_FRAG);
  float* grid     = (float*)(ws + OFF_GRID);
  unsigned* vflag = (unsigned*)(ws + OFF_VALID);
  float* xvx      = (float*)(ws + OFF_XVX);
  float* E0 = (float*)(ws + OFF_E0);
  float* E1 = (float*)(ws + OFF_E1);
  float* E2 = (float*)(ws + OFF_E2);
  float* D0 = (float*)(ws + OFF_D0);
  float* D1 = (float*)(ws + OFF_D1);
  float* D2 = (float*)(ws + OFF_D2);
  float* G  = (float*)(ws + OFF_G);

  float* out = (float*)d_out;
  float* out_valid = out + OUT_VALID;
  float* out_g     = out + OUT_G;
  float* out_feat  = out + OUT_FEAT;
  float* out_aabb  = out + OUT_AABB;
  float* out_orig  = out + OUT_ORIG;

  // 1. init reducers / voxel grid / flags
  init_ws_kernel<<<(NB * VOX * BD + 255) / 256, 256, 0, stream>>>(grid, vflag, red);
  // 2. weight fragments (f32 -> f16, W^T in WMMA A layout, layer0 bias folded)
  prep_frags_kernel<<<(20 * 512 + 255) / 256, 256, 0, stream>>>(w0, b0, w1, w2, frag);
  // 3. per-batch min/max
  {
    dim3 g(64, NB);
    reduce_minmax_kernel<<<g, 256, 0, stream>>>(xyz, red);
  }
  // 4. finalize aabb / params
  finalize_aabb_kernel<<<1, 32, 0, stream>>>(red, ls, params, out_aabb, out_orig);
  // 5. WMMA point MLP + segment-max scatter
  mlp_scatter_kernel<<<256, 256, 0, stream>>>(xyz, normal, frag, b1, b2,
                                              params, grid, vflag);
  // 6. assemble x_vx (B,343,74) and valid output
  voxel_assemble_kernel<<<(NB * VOX * 74 + 255) / 256, 256, 0, stream>>>(
      grid, vflag, params, xvx, out_valid);
  // 7. U-Net (encoder VALID/VALID/SAME, decoder SAME/FULL/FULL, no kernel flip)
  conv3d_ln_kernel<<<NB * 125, 64, 0, stream>>>(xvx, nullptr, 74, 0, 7, 5, 0,
                                                ke0, be0, ge0, he0, E0);
  conv3d_ln_kernel<<<NB * 27, 64, 0, stream>>>(E0, nullptr, 64, 0, 5, 3, 0,
                                               ke1, be1, ge1, he1, E1);
  conv3d_ln_kernel<<<NB * 27, 64, 0, stream>>>(E1, nullptr, 64, 0, 3, 3, 1,
                                               ke2, be2, ge2, he2, E2);
  gmean_kernel<<<NB, 64, 0, stream>>>(E2, G, out_g);
  conv3d_ln_kernel<<<NB * 27, 64, 0, stream>>>(E2, nullptr, 64, 0, 3, 3, 1,
                                               kd0, bd0, gd0, hd0, D0);
  conv3d_ln_kernel<<<NB * 125, 64, 0, stream>>>(D0, E1, 64, 64, 3, 5, 2,
                                                kd1, bd1, gd1, hd1, D1);
  conv3d_ln_kernel<<<NB * VOX, 64, 0, stream>>>(D1, E0, 64, 64, 5, 7, 2,
                                                kd2, bd2, gd2, hd2, D2);
  // 8. final mix + feat
  mix_feat_kernel<<<NB * VOX, 64, 0, stream>>>(D2, G, wm, bm, gm, hm, wo, bo, out_feat);
}
